// CRNNModel_32573031973446
// MI455X (gfx1250) — matrix operand order repaired and checked by reading, not compiled, not run on hardware
//
#include <hip/hip_runtime.h>
#include <hip/hip_bf16.h>
#include <math.h>

typedef __attribute__((ext_vector_type(16))) __bf16 v16bf;
typedef __attribute__((ext_vector_type(8)))  float  v8f;

// global (addrspace 1) views: weight fetches lower to global_load_b128
typedef __bf16 __attribute__((address_space(1))) gbf16;
typedef v16bf  __attribute__((address_space(1))) gv16bf;

#define Bq   1024
#define Nn   256
#define Hh   256
#define TH3  768
#define ROWS_PER_WG 16
#define THREADS 256

#if defined(__HIP_DEVICE_COMPILE__)
#define SCHED_FENCE() __builtin_amdgcn_sched_barrier(0)
#define OPAQUE64(x)   asm volatile("" : "+v"(x))
#else
#define SCHED_FENCE()
#define OPAQUE64(x)
#endif

// ---------------------------------------------------------------------------
// Pack Wh (f32, [256,768] row-major) into bf16 in the WMMA B-tile layout:
//   tiles indexed [kt(8)][ct(48)][lane(32)][e(16)]
// ---------------------------------------------------------------------------
__global__ void pack_wh(const float* __restrict__ Wh, __bf16* __restrict__ whp) {
    int idx = blockIdx.x * blockDim.x + threadIdx.x;   // 0 .. 8*48*32-1
    if (idx >= 8 * 48 * 32) return;
    int kt   = idx / (48 * 32);
    int rem  = idx % (48 * 32);
    int ct   = rem / 32;
    int lane = rem % 32;
    int krow = kt * 32 + ((lane >> 4) << 3);
    const float* src = Wh + krow * TH3 + ct * 16 + (lane & 15);
    __bf16* dst = whp + idx * 16;
#pragma unroll
    for (int e = 0; e < 16; ++e) dst[e] = (__bf16)src[((e & 7) + ((e >> 3) << 4)) * TH3];
}

// ---------------------------------------------------------------------------
// Full GRU scan + readout. One workgroup = 16 batch rows (one M-tile), 8 waves.
// Each wave owns 2 hidden-column tiles (hct = wave*2 + hc) and computes the
// r/z/n gates together.  Grid = 64 workgroups -> 64 WGPs busy.
// ---------------------------------------------------------------------------
__global__ void __launch_bounds__(THREADS, 1) crnn_kernel(
    const int*   __restrict__ inp,   // [B,N] int32
    const float* __restrict__ Wi,    // [2,768]
    const float* __restrict__ bvec,  // [768]
    const float* __restrict__ Wd,    // [256,2]
    const float* __restrict__ bd,    // [2]
    const float* __restrict__ Wp,    // [2,2]
    const float* __restrict__ bp,    // [2]
    const __bf16* __restrict__ whp,  // packed Wh tiles (bf16)
    float* __restrict__ out)         // [B] complex64 -> 2*B floats
{
    __shared__ float hF[ROWS_PER_WG * Hh];     // 16 KB hidden state (f32)
    __shared__ int   xAll[ROWS_PER_WG * Nn];   // 16 KB input slice for this WG
    __shared__ float WdL[Hh * 2];              // 2 KB
    __shared__ float misc[8];                  // Wp(4), bd(2), bp(2)

    const int tid      = threadIdx.x;
    const int lane     = tid & 31;
    const int wave     = tid >> 5;      // 0..7 : hidden-column octant
    const int wgRow0   = blockIdx.x * ROWS_PER_WG;
    const int laneM    = lane & 15;
    const int laneHalf = lane >> 4;

    for (int i = tid; i < ROWS_PER_WG * Nn; i += THREADS) xAll[i] = inp[wgRow0 * Nn + i];
    for (int i = tid; i < Hh * 2;           i += THREADS) WdL[i]  = Wd[i];
    if (tid < 4) misc[tid] = Wp[tid];
    if (tid == 4) { misc[4] = bd[0]; misc[5] = bd[1]; misc[6] = bp[0]; misc[7] = bp[1]; }
    for (int i = tid; i < ROWS_PER_WG * Hh; i += THREADS) hF[i] = 0.0f;

    // --- time-invariant per-lane constants ----------------------------------
    // r,z gates: preactivation = acc + sel, sel in { b, w0+b, w1+b }
    // n gate:    n = tanh(inn + r*(acc + b_n)), inn in { 0, w0, w1 }
    float sR[2][3], sZ[2][3], iN[2][3], bN2[2];
#pragma unroll
    for (int hc = 0; hc < 2; ++hc) {
        const int j = (wave * 2 + hc) * 16 + laneM;
        const float br_ = bvec[j], bz_ = bvec[256 + j], bn_ = bvec[512 + j];
        sR[hc][0] = br_;  sR[hc][1] = Wi[j]       + br_; sR[hc][2] = Wi[768 + j]  + br_;
        sZ[hc][0] = bz_;  sZ[hc][1] = Wi[256 + j] + bz_; sZ[hc][2] = Wi[1024 + j] + bz_;
        iN[hc][0] = 0.0f; iN[hc][1] = Wi[512 + j];       iN[hc][2] = Wi[1280 + j];
        bN2[hc]   = bn_;
    }

    // base of this lane's B tiles; tile offsets are compile-time immediates:
    //   bytes: lane*32 + wave*2048  (+ hc*1024 + gate*16384 + k*49152)
    unsigned long long bbu =
        (unsigned long long)(whp + (size_t)(lane * 16 + wave * 1024));

    float sumL = 0.0f, sumP = 0.0f;   // valid in threads with (tid&15)==0

    for (int t = 0; t < Nn; ++t) {
        __syncthreads();   // (1) prev-step h writes + readout reads complete

        // defeat LICM, then re-assert global address space
        OPAQUE64(bbu);
        const gv16bf* P = (const gv16bf*)bbu;   // units of 32B tiles-chunks

        // --- A tile set: 16 batch rows, full K=256, f32->bf16 -------------
        v16bf a[8];
        {
            const float* hrow = &hF[laneM * Hh];
            const int kb = laneHalf * 8;
#pragma unroll
            for (int k = 0; k < 8; ++k) {
#pragma unroll
                for (int e = 0; e < 8; ++e) {
                    a[k][e]     = (__bf16)hrow[k * 32 + kb + e];
                    a[k][e + 8] = (__bf16)hrow[k * 32 + kb + 16 + e];
                }
            }
        }

        // teacher-forced symbol for my 8 D-tile rows (same across hc tiles)
        int x8[8];
#pragma unroll
        for (int v = 0; v < 8; ++v)
            x8[v] = (t > 0) ? xAll[(v + laneHalf * 8) * Nn + (t - 1)] : -1;

        __syncthreads();   // (2) all A reads done before h is overwritten
        SCHED_FENCE();

#pragma unroll
        for (int hc = 0; hc < 2; ++hc) {
            // gv16bf strides: hc -> 32, gate -> 512, k -> 1536
            v8f accR = {}, accZ = {}, accN = {};   // inline-0 C
            v16bf br = P[hc * 32 + 0];
            v16bf bz = P[hc * 32 + 512];
            v16bf bn = P[hc * 32 + 1024];
#pragma unroll
            for (int k = 0; k < 8; ++k) {
                v16bf nbr, nbz, nbn;                 // 1-deep prefetch pipeline
                if (k < 7) {
                    nbr = P[hc * 32 + (k + 1) * 1536 + 0];
                    nbz = P[hc * 32 + (k + 1) * 1536 + 512];
                    nbn = P[hc * 32 + (k + 1) * 1536 + 1024];
                }
                accR = __builtin_amdgcn_wmma_f32_16x16x32_bf16(false, a[k], false, br, (short)0, accR, false, false);
                accZ = __builtin_amdgcn_wmma_f32_16x16x32_bf16(false, a[k], false, bz, (short)0, accZ, false, false);
                accN = __builtin_amdgcn_wmma_f32_16x16x32_bf16(false, a[k], false, bn, (short)0, accN, false, false);
                SCHED_FENCE();   // keep in-flight window to ~2 k-steps
                if (k < 7) { br = nbr; bz = nbz; bn = nbn; }
            }

            // --- gate math straight out of accumulators ---------------------
#pragma unroll
            for (int v = 0; v < 8; ++v) {
                const int   row = v + laneHalf * 8;
                const int   s   = x8[v] + 1;           // 0:none, 1:sym0, 2:sym1
                const float pr  = (s == 0) ? sR[hc][0] : ((s == 1) ? sR[hc][1] : sR[hc][2]);
                const float pz  = (s == 0) ? sZ[hc][0] : ((s == 1) ? sZ[hc][1] : sZ[hc][2]);
                const float inn = (s == 0) ? iN[hc][0] : ((s == 1) ? iN[hc][1] : iN[hc][2]);
                const float r   = 1.0f / (1.0f + __expf(-(accR[v] + pr)));
                const float z   = 1.0f / (1.0f + __expf(-(accZ[v] + pz)));
                const float n   = tanhf(inn + r * (accN[v] + bN2[hc]));
                const int   idx = row * Hh + (wave * 2 + hc) * 16 + laneM;
                const float ho  = hF[idx];
                hF[idx] = (1.0f - z) * n + z * ho;
            }
        }
        __syncthreads();   // (3) h_t complete for all 256 hidden units

        // --- readout: x = h@Wd + bd ; D=2 log-softmax + phases, gather ------
        {
            const int r   = tid >> 4;      // row within WG (0..15)
            const int seg = tid & 15;      // 16 threads cooperate per row
            const float* hrow = &hF[r * Hh];
            float s0 = 0.0f, s1 = 0.0f;
#pragma unroll 4
            for (int jj = seg * 16; jj < seg * 16 + 16; ++jj) {
                const float hv = hrow[jj];
                s0 += hv * WdL[jj * 2 + 0];
                s1 += hv * WdL[jj * 2 + 1];
            }
            s0 += __shfl_down(s0, 8); s1 += __shfl_down(s1, 8);
            s0 += __shfl_down(s0, 4); s1 += __shfl_down(s1, 4);
            s0 += __shfl_down(s0, 2); s1 += __shfl_down(s1, 2);
            s0 += __shfl_down(s0, 1); s1 += __shfl_down(s1, 1);
            if (seg == 0) {
                const float x0  = s0 + misc[4];
                const float x1  = s1 + misc[5];
                const int   tgt = xAll[r * Nn + t];
                const float mx  = fmaxf(x0, x1);
                const float lse = mx + __logf(__expf(x0 - mx) + __expf(x1 - mx));
                const float lg  = ((tgt == 0) ? x0 : x1) - lse;
                const float p0  = x0 * misc[0] + x1 * misc[2] + misc[6];
                const float p1  = x0 * misc[1] + x1 * misc[3] + misc[7];
                const float pd  = (tgt == 0) ? p0 : p1;
                sumL += lg;
                sumP += 3.14159265358979323846f * (pd / (1.0f + fabsf(pd)));
            }
        }
    }

    if ((tid & 15) == 0) {
        const int b = wgRow0 + (tid >> 4);
        out[2 * b + 0] = 0.5f * sumL;   // real part
        out[2 * b + 1] = sumP;          // imag part
    }
}

extern "C" void kernel_launch(void* const* d_in, const int* in_sizes, int n_in,
                              void* d_out, int out_size, void* d_ws, size_t ws_size,
                              hipStream_t stream) {
    const int*   inp = (const int*)  d_in[0];
    const float* Wi  = (const float*)d_in[1];
    const float* Wh  = (const float*)d_in[2];
    const float* bv  = (const float*)d_in[3];
    const float* Wd  = (const float*)d_in[4];
    const float* bd  = (const float*)d_in[5];
    const float* Wp  = (const float*)d_in[6];
    const float* bp  = (const float*)d_in[7];
    float*  out = (float*)d_out;
    __bf16* whp = (__bf16*)d_ws;      // 8*48*32*16 bf16 = 384 KB of scratch

    pack_wh<<<48, 256, 0, stream>>>(Wh, whp);
    crnn_kernel<<<Bq / ROWS_PER_WG, THREADS, 0, stream>>>(
        inp, Wi, bv, Wd, bd, Wp, bp, whp, out);
}
